// PolicyNetwork3_84593675862715
// MI455X (gfx1250) — matrix operand order repaired and checked by reading, not compiled, not run on hardware
//
#include <hip/hip_runtime.h>
#include <hip/hip_bf16.h>
#include <math.h>

typedef __attribute__((ext_vector_type(16))) _Float16 v16h;
typedef __attribute__((ext_vector_type(8)))  float    v8f;
typedef __attribute__((__vector_size__(4 * sizeof(int)))) int gv4i;

#define N_NODES 50000
#define N_EDGES 800000
#define N_CAND  100000
#define DHID    128
#define NH      (N_NODES * DHID)
#define NEG_SLOPE 0.01f
#define BN_EPS    1e-5f
#define RED_BLOCKS 512

// ---------------- WMMA helpers (gfx1250, wave32) ----------------

__device__ __forceinline__ v8f wmma16(v16h a, v16h b, v8f c) {
  // D = A(16x32 f16) * B(32x16 f16) + C(16x16 f32)
  return __builtin_amdgcn_wmma_f32_16x16x32_f16(false, a, false, b, (short)0, c,
                                                false, false);
}

// A fragment: 16x32 f16, row-major source with leading dim ld.
// Per ISA 7.12.2: lanes 0-15 hold K {k0..k0+7, k0+16..k0+23} of row M=lane,
// lanes 16-31 hold K {k0+8..k0+15, k0+24..k0+31} of row M=lane-16.
// 'row' is the absolute per-lane row index (caller applies lane&15).
__device__ __forceinline__ v16h load_a16(const _Float16* base, int ld, int row,
                                         int k0, int lane) {
  const _Float16* p = base + (size_t)row * ld + (k0 + ((lane >> 4) << 3));
  v16h a;
#pragma unroll
  for (int e = 0; e < 8; ++e) { a[e] = p[e]; a[e + 8] = p[e + 16]; }
  return a;
}

// B fragment: 32x16 f16 from TRANSPOSED weights Wt[ncols][K] (ld = K).
// lanes 0-15: column n0+lane, K {k0..k0+7, k0+16..k0+23}; lanes 16-31: K {k0+8..k0+15, k0+24..k0+31}.
__device__ __forceinline__ v16h load_bT(const _Float16* wt, int ldk, int k0,
                                        int n0, int lane) {
  const _Float16* p = wt + (size_t)(n0 + (lane & 15)) * ldk
                         + (k0 + ((lane >> 4) << 3));
  v16h b;
#pragma unroll
  for (int e = 0; e < 8; ++e) { b[e] = p[e]; b[e + 8] = p[e + 16]; }
  return b;
}

// ---------------- async global->LDS (gfx1250 ASYNCcnt path) ----------------

__device__ __forceinline__ void async_copy16(const _Float16* g, _Float16* l) {
#if defined(__gfx1250__) && __has_builtin(__builtin_amdgcn_global_load_async_to_lds_b128)
  __builtin_amdgcn_global_load_async_to_lds_b128(
      (__attribute__((address_space(1))) gv4i*)const_cast<_Float16*>(g),
      (__attribute__((address_space(3))) gv4i*)l, 0, 0);
#else
  ((int4*)l)[0] = ((const int4*)g)[0];
#endif
}

__device__ __forceinline__ void wait_async() {
#if defined(__gfx1250__)
#if __has_builtin(__builtin_amdgcn_s_wait_asynccnt)
  __builtin_amdgcn_s_wait_asynccnt(0);
#else
  asm volatile("s_wait_asynccnt 0x0" ::: "memory");
#endif
#endif
}

// ---------------- prep / conversion kernels ----------------

__global__ void k_prep(const float* ws0, const float* wn0, const float* ws1,
                       const float* wn1, const float* mw0, const float* mw1,
                       const float* b0, const float* g0, const float* be0,
                       const float* rm0, const float* rv0,
                       const float* b1, const float* g1, const float* be1,
                       const float* rm1, const float* rv1,
                       _Float16* ws0t, _Float16* wn0t, _Float16* ws1t,
                       _Float16* wn1t, _Float16* mw0t, _Float16* mw1t,
                       float* scale0, float* bias0, float* scale1, float* bias1) {
  int i = blockIdx.x * blockDim.x + threadIdx.x;
  if (i < DHID * DHID) {                    // transpose + f16 all 4 conv weights
    int k = i / DHID, c = i % DHID;
    int t = c * DHID + k;
    ws0t[t] = (_Float16)ws0[i];
    wn0t[t] = (_Float16)wn0[i];
    ws1t[t] = (_Float16)ws1[i];
    wn1t[t] = (_Float16)wn1[i];
  }
  if (i < 256 * 64) {                       // mw0 rows 0..255 -> mw0t[64][256]
    int k = i / 64, c = i % 64;
    mw0t[c * 256 + k] = (_Float16)mw0[i];
  }
  if (i < 64 * 64) {                        // mw1 -> mw1t[64][64]
    int k = i / 64, c = i % 64;
    mw1t[c * 64 + k] = (_Float16)mw1[i];
  }
  if (i < DHID) {                           // fold bias + BN: y = gemm*scale + bias2
    float s0 = g0[i] * rsqrtf(rv0[i] + BN_EPS);
    scale0[i] = s0;
    bias0[i]  = (b0[i] - rm0[i]) * s0 + be0[i];
    float s1 = g1[i] * rsqrtf(rv1[i] + BN_EPS);
    scale1[i] = s1;
    bias1[i]  = (b1[i] - rm1[i]) * s1 + be1[i];
  }
}

__global__ void k_f32_to_f16(const float* in, _Float16* out, int n) {
  int i = blockIdx.x * blockDim.x + threadIdx.x;
  if (i < n) out[i] = (_Float16)in[i];
}

__global__ void k_zero_f32(float* p, int n) {
  int i = blockIdx.x * blockDim.x + threadIdx.x;
  if (i < n) p[i] = 0.0f;
}

// ---------------- graph aggregation ----------------

__global__ void k_degree(const int* dst, float* deg, int e) {
  int i = blockIdx.x * blockDim.x + threadIdx.x;
  if (i < e) atomicAdd(&deg[dst[i]], 1.0f);
}

// one wave per edge; lane covers 4 channels
__global__ void k_scatter(const _Float16* h, const int* src, const int* dst,
                          float* agg, int e) {
  int w = (blockIdx.x * blockDim.x + threadIdx.x) >> 5;
  int lane = threadIdx.x & 31;
  if (w >= e) return;
  int s = src[w], d = dst[w];
  const _Float16* p = h + (size_t)s * DHID + lane * 4;
  float* q = agg + (size_t)d * DHID + lane * 4;
#pragma unroll
  for (int i = 0; i < 4; ++i) atomicAdd(q + i, (float)p[i]);
}

__global__ void k_normalize(const float* agg, const float* deg, _Float16* aggn,
                            int n) {
  int i = blockIdx.x * blockDim.x + threadIdx.x;
  if (i < n) {
    int node = i >> 7;
    aggn[i] = (_Float16)(agg[i] / fmaxf(deg[node], 1.0f));
  }
}

// ---------------- fused SAGE GEMM: H = leaky((A*Ws + G*Wn)*scale + bias2) ----
// grid = N/16 blocks x 256 threads (8 waves = 8 column tiles of 16).
// A-tiles (x rows + aggn rows, 8KB) staged once per block into LDS via the
// gfx1250 async global->LDS path, then all 8 waves feed WMMA from LDS.

__global__ __launch_bounds__(256) void k_sage(const _Float16* __restrict__ A,
                                              const _Float16* __restrict__ G,
                                              const _Float16* __restrict__ wsT,
                                              const _Float16* __restrict__ wnT,
                                              const float* __restrict__ scale,
                                              const float* __restrict__ bias,
                                              _Float16* __restrict__ H,
                                              int fixnan) {
  __shared__ _Float16 tA[16][136];   // 272B row stride: avoids bank aliasing
  __shared__ _Float16 tG[16][136];
  int tt = threadIdx.x;
  int row0 = blockIdx.x * 16;
  {
    int tr = tt >> 4, tseg = tt & 15;   // 256 threads x 16B per tile
    async_copy16(A + (size_t)(row0 + tr) * DHID + tseg * 8, &tA[tr][tseg * 8]);
    async_copy16(G + (size_t)(row0 + tr) * DHID + tseg * 8, &tG[tr][tseg * 8]);
    wait_async();
  }
  __syncthreads();

  int wv = tt >> 5;                   // column tile 0..7
  int lane = tt & 31;
  int col0 = wv * 16;
  int rlo = lane & 15, hi = lane >> 4;

  v8f acc = {};
#pragma unroll
  for (int kb = 0; kb < DHID; kb += 32) {
    v16h a1 = load_a16(&tA[0][0], 136, rlo, kb, lane);
    v16h b1 = load_bT(wsT, DHID, kb, col0, lane);
    acc = wmma16(a1, b1, acc);
    v16h a2 = load_a16(&tG[0][0], 136, rlo, kb, lane);
    v16h b2 = load_bT(wnT, DHID, kb, col0, lane);
    acc = wmma16(a2, b2, acc);
  }

  int col = col0 + rlo;
  float sc = scale[col], bi = bias[col];
  int rbase = row0 + (hi << 3);
#pragma unroll
  for (int r = 0; r < 8; ++r) {
    float v = acc[r] * sc + bi;
    v = v > 0.0f ? v : v * NEG_SLOPE;
    if (fixnan && (v != v)) v = 1e-14f;   // nan_to_num
    H[(size_t)(rbase + r) * DHID + col] = (_Float16)v;
  }
}

// ---------------- candidate MLP: 257 -> 64 -> 64 -> 1 ----------------
// grid = 625 blocks x 320 threads (10 waves); wave handles 16 candidates.

__global__ __launch_bounds__(320) void k_mlp(
    const _Float16* __restrict__ H, const int* __restrict__ cu,
    const int* __restrict__ cv, const float* __restrict__ cf,
    const _Float16* __restrict__ m0t, const float* __restrict__ mw0last,
    const float* __restrict__ mb0, const _Float16* __restrict__ m1t,
    const float* __restrict__ mb1, const float* __restrict__ mw2,
    const float* __restrict__ mb2, float* __restrict__ y) {
  __shared__ _Float16 zs[10][16][72];
  int wv = threadIdx.x >> 5;
  int lane = threadIdx.x & 31;
  int tile = blockIdx.x * 10 + wv;         // 625*10 == 6250 == C/16 exactly
  int rlo = lane & 15, hi = lane >> 4;
  int rowA = tile * 16 + rlo;
  int u = cu[rowA], v = cv[rowA];

  // layer 0: z = leaky(h[u]@W0[0:128] + h[v]@W0[128:256] + cf*W0[256] + b0)
  v8f acc[4] = {};
#pragma unroll
  for (int kb = 0; kb < DHID; kb += 32) {
    v16h au = load_a16(H, DHID, u, kb, lane);
#pragma unroll
    for (int t = 0; t < 4; ++t)
      acc[t] = wmma16(au, load_bT(m0t, 256, kb, t * 16, lane), acc[t]);
    v16h av = load_a16(H, DHID, v, kb, lane);
#pragma unroll
    for (int t = 0; t < 4; ++t)
      acc[t] = wmma16(av, load_bT(m0t, 256, 128 + kb, t * 16, lane), acc[t]);
  }
  float cfr[8];
  int rbase = tile * 16 + (hi << 3);
#pragma unroll
  for (int r = 0; r < 8; ++r) cfr[r] = cf[rbase + r];
#pragma unroll
  for (int t = 0; t < 4; ++t) {
    int col = t * 16 + rlo;
    float wl = mw0last[col], bb = mb0[col];
#pragma unroll
    for (int r = 0; r < 8; ++r) {
      float val = acc[t][r] + cfr[r] * wl + bb;
      val = val > 0.0f ? val : val * NEG_SLOPE;
      zs[wv][(hi << 3) + r][col] = (_Float16)val;
    }
  }
  __syncthreads();

  // layer 1: z2 = leaky(z @ mw1 + mb1)
  v8f bcc[4] = {};
#pragma unroll
  for (int kb = 0; kb < 64; kb += 32) {
    v16h az = load_a16(&zs[wv][0][0], 72, rlo, kb, lane);
#pragma unroll
    for (int t = 0; t < 4; ++t)
      bcc[t] = wmma16(az, load_bT(m1t, 64, kb, t * 16, lane), bcc[t]);
  }
  __syncthreads();
#pragma unroll
  for (int t = 0; t < 4; ++t) {
    int col = t * 16 + rlo;
    float bb = mb1[col];
#pragma unroll
    for (int r = 0; r < 8; ++r) {
      float val = bcc[t][r] + bb;
      val = val > 0.0f ? val : val * NEG_SLOPE;
      zs[wv][(hi << 3) + r][col] = (_Float16)val;
    }
  }
  __syncthreads();

  // layer 2: y = z2 @ mw2 + mb2  (64-dot per row)
  if (lane < 16) {
    float sum = mb2[0];
#pragma unroll
    for (int k = 0; k < 64; ++k) sum += (float)zs[wv][lane][k] * mw2[k];
    y[tile * 16 + lane] = sum;
  }
}

// ---------------- deterministic softmax over C ----------------

__global__ void k_partial_max(const float* y, int n, float* part) {
  __shared__ float s[256];
  float m = -1e30f;
  for (int i = blockIdx.x * 256 + threadIdx.x; i < n; i += gridDim.x * 256)
    m = fmaxf(m, y[i]);
  s[threadIdx.x] = m;
  __syncthreads();
  for (int o = 128; o; o >>= 1) {
    if (threadIdx.x < o) s[threadIdx.x] = fmaxf(s[threadIdx.x], s[threadIdx.x + o]);
    __syncthreads();
  }
  if (threadIdx.x == 0) part[blockIdx.x] = s[0];
}

__global__ void k_partial_expsum(const float* y, int n, const float* gmax,
                                 float* part) {
  __shared__ float s[256];
  float gm = *gmax;
  float acc = 0.0f;
  for (int i = blockIdx.x * 256 + threadIdx.x; i < n; i += gridDim.x * 256)
    acc += expf(y[i] - gm);
  s[threadIdx.x] = acc;
  __syncthreads();
  for (int o = 128; o; o >>= 1) {
    if (threadIdx.x < o) s[threadIdx.x] += s[threadIdx.x + o];
    __syncthreads();
  }
  if (threadIdx.x == 0) part[blockIdx.x] = s[0];
}

__global__ void k_final_reduce(const float* part, int nb, float* out, int isMax) {
  __shared__ float s[256];
  float v = isMax ? -1e30f : 0.0f;
  for (int i = threadIdx.x; i < nb; i += 256)
    v = isMax ? fmaxf(v, part[i]) : v + part[i];
  s[threadIdx.x] = v;
  __syncthreads();
  for (int o = 128; o; o >>= 1) {
    if (threadIdx.x < o)
      s[threadIdx.x] = isMax ? fmaxf(s[threadIdx.x], s[threadIdx.x + o])
                             : s[threadIdx.x] + s[threadIdx.x + o];
    __syncthreads();
  }
  if (threadIdx.x == 0) out[0] = s[0];
}

__global__ void k_write_softmax(const float* y, int n, const float* gmax,
                                const float* gsum, float* out) {
  int i = blockIdx.x * blockDim.x + threadIdx.x;
  if (i < n) out[i] = expf(y[i] - *gmax) / *gsum;
}

// ---------------- host-side orchestration ----------------

extern "C" void kernel_launch(void* const* d_in, const int* in_sizes, int n_in,
                              void* d_out, int out_size, void* d_ws,
                              size_t ws_size, hipStream_t stream) {
  const float* x   = (const float*)d_in[0];
  const int* src   = (const int*)d_in[1];
  const int* dst   = (const int*)d_in[2];
  const int* cu    = (const int*)d_in[3];
  const int* cv    = (const int*)d_in[4];
  const float* cf  = (const float*)d_in[5];
  const float* ws0 = (const float*)d_in[6];
  const float* wn0 = (const float*)d_in[7];
  const float* b0  = (const float*)d_in[8];
  const float* g0  = (const float*)d_in[9];
  const float* be0 = (const float*)d_in[10];
  const float* rm0 = (const float*)d_in[11];
  const float* rv0 = (const float*)d_in[12];
  const float* ws1 = (const float*)d_in[13];
  const float* wn1 = (const float*)d_in[14];
  const float* b1  = (const float*)d_in[15];
  const float* g1  = (const float*)d_in[16];
  const float* be1 = (const float*)d_in[17];
  const float* rm1 = (const float*)d_in[18];
  const float* rv1 = (const float*)d_in[19];
  const float* mw0 = (const float*)d_in[20];
  const float* mb0 = (const float*)d_in[21];
  const float* mw1 = (const float*)d_in[22];
  const float* mb1 = (const float*)d_in[23];
  const float* mw2 = (const float*)d_in[24];
  const float* mb2 = (const float*)d_in[25];

  char* base = (char*)d_ws;
  size_t off = 0;
  auto alloc = [&](size_t bytes) -> void* {
    void* p = base + off;
    off = (off + bytes + 255) & ~(size_t)255;
    return p;
  };

  _Float16* xh   = (_Float16*)alloc((size_t)NH * 2);
  _Float16* h0   = (_Float16*)alloc((size_t)NH * 2);
  _Float16* h1   = (_Float16*)alloc((size_t)NH * 2);
  float*    agg  = (float*)alloc((size_t)NH * 4);
  _Float16* aggn = (_Float16*)alloc((size_t)NH * 2);
  float*    deg  = (float*)alloc((size_t)N_NODES * 4);
  _Float16* ws0t = (_Float16*)alloc(DHID * DHID * 2);
  _Float16* wn0t = (_Float16*)alloc(DHID * DHID * 2);
  _Float16* ws1t = (_Float16*)alloc(DHID * DHID * 2);
  _Float16* wn1t = (_Float16*)alloc(DHID * DHID * 2);
  _Float16* mw0t = (_Float16*)alloc(256 * 64 * 2);
  _Float16* mw1t = (_Float16*)alloc(64 * 64 * 2);
  float* scale0  = (float*)alloc(DHID * 4);
  float* bias0   = (float*)alloc(DHID * 4);
  float* scale1  = (float*)alloc(DHID * 4);
  float* bias1   = (float*)alloc(DHID * 4);
  float* part    = (float*)alloc(RED_BLOCKS * 4);
  float* gred    = (float*)alloc(8);   // [0]=gmax [1]=gsum

  k_prep<<<64, 256, 0, stream>>>(ws0, wn0, ws1, wn1, mw0, mw1,
                                 b0, g0, be0, rm0, rv0,
                                 b1, g1, be1, rm1, rv1,
                                 ws0t, wn0t, ws1t, wn1t, mw0t, mw1t,
                                 scale0, bias0, scale1, bias1);
  k_f32_to_f16<<<(NH + 255) / 256, 256, 0, stream>>>(x, xh, NH);
  k_zero_f32<<<(NH + 255) / 256, 256, 0, stream>>>(agg, NH);
  k_zero_f32<<<(N_NODES + 255) / 256, 256, 0, stream>>>(deg, N_NODES);
  k_degree<<<(N_EDGES + 255) / 256, 256, 0, stream>>>(dst, deg, N_EDGES);

  // ---- layer 0 ----
  k_scatter<<<N_EDGES / 8, 256, 0, stream>>>(xh, src, dst, agg, N_EDGES);
  k_normalize<<<(NH + 255) / 256, 256, 0, stream>>>(agg, deg, aggn, NH);
  k_sage<<<N_NODES / 16, 256, 0, stream>>>(xh, aggn, ws0t, wn0t, scale0, bias0,
                                           h0, 0);
  // ---- layer 1 ----
  k_zero_f32<<<(NH + 255) / 256, 256, 0, stream>>>(agg, NH);
  k_scatter<<<N_EDGES / 8, 256, 0, stream>>>(h0, src, dst, agg, N_EDGES);
  k_normalize<<<(NH + 255) / 256, 256, 0, stream>>>(agg, deg, aggn, NH);
  k_sage<<<N_NODES / 16, 256, 0, stream>>>(h0, aggn, ws1t, wn1t, scale1, bias1,
                                           h1, 1);

  // ---- candidate MLP ----
  float* y  = (float*)d_out;
  float* sm = y + N_CAND;
  k_mlp<<<625, 320, 0, stream>>>(h1, cu, cv, cf, mw0t, mw0 + 256 * 64, mb0,
                                 mw1t, mb1, mw2, mb2, y);

  // ---- softmax over C (deterministic two-pass) ----
  k_partial_max<<<RED_BLOCKS, 256, 0, stream>>>(y, N_CAND, part);
  k_final_reduce<<<1, 256, 0, stream>>>(part, RED_BLOCKS, gred + 0, 1);
  k_partial_expsum<<<RED_BLOCKS, 256, 0, stream>>>(y, N_CAND, gred + 0, part);
  k_final_reduce<<<1, 256, 0, stream>>>(part, RED_BLOCKS, gred + 1, 0);
  k_write_softmax<<<(N_CAND + 255) / 256, 256, 0, stream>>>(y, N_CAND, gred + 0,
                                                            gred + 1, sm);
}
